// MSRNN_21088289423705
// MI455X (gfx1250) — matrix operand order repaired and checked by reading, not compiled, not run on hardware
//
#include <hip/hip_runtime.h>
#include <hip/hip_bf16.h>

// ---------------------------------------------------------------------------
// MSRNN (3-layer multi-scale GRU) for gfx1250 using v_wmma_f32_16x16x32_f16.
// N_INPUT=256, N_HIDDEN=1024, BATCH=32, SEQ=512.
// Ping-pong software-pipelined K loops; per-lane base pointers advance by a
// constant 128B per iteration so all loads use immediate offsets (no per-
// iteration 64-bit address rebuilds, lower register pressure).
// ---------------------------------------------------------------------------

typedef _Float16 f16;
typedef __attribute__((ext_vector_type(16))) _Float16 v16h;
typedef __attribute__((ext_vector_type(8)))  _Float16 v8h;
typedef __attribute__((ext_vector_type(8)))  float    v8f;

union AFrag { v16h v; v8h h[2]; };

// A-matrix (16x32 f16, M x K): per-lane pointer p = &A[rowM + lane15][half*16].
// Fragment at k0: elems 0-7 at p+k0, elems 8-15 at p+k0+8.
__device__ __forceinline__ v16h load_a_at(const f16* __restrict__ p, int k0) {
  AFrag a;
  a.h[0] = *(const v8h*)(p + k0);
  a.h[1] = *(const v8h*)(p + k0 + 8);
  return a.v;
}

// B-matrix (32x16 f16, K x N) from row-major W[N][K]: per-lane pointer
// p = &W[colN + lane15][half*16].  Fragment at k0: p+k0 and p+k0+8.
__device__ __forceinline__ v16h load_b_at(const f16* __restrict__ p, int k0) {
  AFrag b;
  b.h[0] = *(const v8h*)(p + k0);
  b.h[1] = *(const v8h*)(p + k0 + 8);
  return b.v;
}

__device__ __forceinline__ v8f wmma_f16(v16h a, v16h b, v8f c) {
  return __builtin_amdgcn_wmma_f32_16x16x32_f16(false, a, false, b,
                                                (short)0, c, false, false);
}

// ---------------------------------------------------------------------------
// Generic GEMM: C[M x N] (f32) = A[M x K](f16) * W[N x K](f16)^T (+bias1+bias2)
// optionally accumulating into existing C.  Workgroup tile: 32(M) x 256(N),
// 8 waves, each wave computes a 16 x 64 strip (A fragment reused 4x).
// grid = (N/256, M/32), block = 256.  Requires K % 64 == 0 (even step count).
// ---------------------------------------------------------------------------
__global__ __launch_bounds__(256)
void msrnn_gemm_f16(const f16* __restrict__ A, int lda,
                    const f16* __restrict__ W, int K,
                    const float* __restrict__ bias1,
                    const float* __restrict__ bias2,
                    float* __restrict__ C, int ldc,
                    int accumulate) {
  int lane   = threadIdx.x & 31;
  int lane15 = lane & 15;
  int half   = lane >> 4;
  int w      = threadIdx.x >> 5;
  int mt     = w & 1;
  int nq     = w >> 1;
  int rowM    = blockIdx.y * 32 + mt * 16;
  int colBase = blockIdx.x * 256 + nq * 64;

  // per-lane base pointers; advanced by 64 elements (2 k-steps) per iteration
  const f16* ap  = A + (size_t)(rowM + lane15) * lda + half * 16;
  const f16* bp0 = W + (size_t)(colBase +  0 + lane15) * K + half * 16;
  const f16* bp1 = W + (size_t)(colBase + 16 + lane15) * K + half * 16;
  const f16* bp2 = W + (size_t)(colBase + 32 + lane15) * K + half * 16;
  const f16* bp3 = W + (size_t)(colBase + 48 + lane15) * K + half * 16;

  v8f acc[4] = {};

  // ping-pong pipelined K loop, S = K/32 steps (S even)
  v16h a0 = load_a_at(ap, 0);
  v16h b00 = load_b_at(bp0, 0), b01 = load_b_at(bp1, 0),
       b02 = load_b_at(bp2, 0), b03 = load_b_at(bp3, 0);
  v16h a1, b10, b11, b12, b13;

  const int S = K >> 5;
  for (int s = 0; s < S - 2; s += 2) {
    a1  = load_a_at(ap, 32);
    b10 = load_b_at(bp0, 32); b11 = load_b_at(bp1, 32);
    b12 = load_b_at(bp2, 32); b13 = load_b_at(bp3, 32);
    acc[0] = wmma_f16(a0, b00, acc[0]);
    acc[1] = wmma_f16(a0, b01, acc[1]);
    acc[2] = wmma_f16(a0, b02, acc[2]);
    acc[3] = wmma_f16(a0, b03, acc[3]);
    a0  = load_a_at(ap, 64);
    b00 = load_b_at(bp0, 64); b01 = load_b_at(bp1, 64);
    b02 = load_b_at(bp2, 64); b03 = load_b_at(bp3, 64);
    acc[0] = wmma_f16(a1, b10, acc[0]);
    acc[1] = wmma_f16(a1, b11, acc[1]);
    acc[2] = wmma_f16(a1, b12, acc[2]);
    acc[3] = wmma_f16(a1, b13, acc[3]);
    ap += 64; bp0 += 64; bp1 += 64; bp2 += 64; bp3 += 64;
  }
  {
    // base now at step S-2 (held in buf0); last step S-1 at +32
    a1  = load_a_at(ap, 32);
    b10 = load_b_at(bp0, 32); b11 = load_b_at(bp1, 32);
    b12 = load_b_at(bp2, 32); b13 = load_b_at(bp3, 32);
    acc[0] = wmma_f16(a0, b00, acc[0]);
    acc[1] = wmma_f16(a0, b01, acc[1]);
    acc[2] = wmma_f16(a0, b02, acc[2]);
    acc[3] = wmma_f16(a0, b03, acc[3]);
    acc[0] = wmma_f16(a1, b10, acc[0]);
    acc[1] = wmma_f16(a1, b11, acc[1]);
    acc[2] = wmma_f16(a1, b12, acc[2]);
    acc[3] = wmma_f16(a1, b13, acc[3]);
  }

#pragma unroll
  for (int j = 0; j < 4; ++j) {
    int col = colBase + j * 16 + lane15;
    float badd = 0.f;
    if (bias1) badd += bias1[col];
    if (bias2) badd += bias2[col];
#pragma unroll
    for (int v = 0; v < 8; ++v) {
      int row = rowM + half * 8 + v;
      size_t idx = (size_t)row * ldc + col;
      float o = acc[j][v] + badd;
      if (accumulate) o += C[idx];
      C[idx] = o;
    }
  }
}

// ---------------------------------------------------------------------------
// One GRU scan step:  gh = h @ Whh^T + bhh  (M=32, N=3072, K=1024), fused with
// gate nonlinearities + state update.  Each workgroup owns 64 hidden units u
// and computes the r/i/n columns for those units, so the full gate math is
// local.  h(f16) is double-buffered across steps (in/out pointers).
// grid = 16 blocks, block = 256 threads (8 waves; wave -> (mt, ut) tile, 3 gates).
// ---------------------------------------------------------------------------
__global__ __launch_bounds__(256)
void msrnn_gru_step(const f16* __restrict__ h16_in,   // [32][1024]
                    const f16* __restrict__ Whh,      // [3072][1024] f16
                    const float* __restrict__ bhh,    // [3072]
                    const float* __restrict__ gsum_t, // [32][3072] precomputed gx+gn+biases
                    float* __restrict__ h32,          // [32][1024] f32 state (in-place)
                    f16* __restrict__ h16_out,        // [32][1024]
                    f16* __restrict__ outs_t)         // [32][1024] layer output (f16)
{
  int lane   = threadIdx.x & 31;
  int lane15 = lane & 15;
  int half   = lane >> 4;
  int w      = threadIdx.x >> 5;
  int mt     = w & 1;
  int ut     = w >> 1;
  int rowM   = mt * 16;
  int u16    = blockIdx.x * 64 + ut * 16;

  const f16* ap  = h16_in + (size_t)(rowM + lane15) * 1024 + half * 16;
  const f16* bpr = Whh + (size_t)(       u16 + lane15) * 1024 + half * 16;
  const f16* bpi = Whh + (size_t)(1024 + u16 + lane15) * 1024 + half * 16;
  const f16* bpn = Whh + (size_t)(2048 + u16 + lane15) * 1024 + half * 16;

  v8f acc[3] = {};

  v16h a0 = load_a_at(ap, 0);
  v16h b0r = load_b_at(bpr, 0), b0i = load_b_at(bpi, 0), b0n = load_b_at(bpn, 0);
  v16h a1, b1r, b1i, b1n;

  for (int s = 0; s < 30; s += 2) {
    a1  = load_a_at(ap, 32);
    b1r = load_b_at(bpr, 32); b1i = load_b_at(bpi, 32); b1n = load_b_at(bpn, 32);
    acc[0] = wmma_f16(a0, b0r, acc[0]);
    acc[1] = wmma_f16(a0, b0i, acc[1]);
    acc[2] = wmma_f16(a0, b0n, acc[2]);
    a0  = load_a_at(ap, 64);
    b0r = load_b_at(bpr, 64); b0i = load_b_at(bpi, 64); b0n = load_b_at(bpn, 64);
    acc[0] = wmma_f16(a1, b1r, acc[0]);
    acc[1] = wmma_f16(a1, b1i, acc[1]);
    acc[2] = wmma_f16(a1, b1n, acc[2]);
    ap += 64; bpr += 64; bpi += 64; bpn += 64;
  }
  {
    a1  = load_a_at(ap, 32);
    b1r = load_b_at(bpr, 32); b1i = load_b_at(bpi, 32); b1n = load_b_at(bpn, 32);
    acc[0] = wmma_f16(a0, b0r, acc[0]);
    acc[1] = wmma_f16(a0, b0i, acc[1]);
    acc[2] = wmma_f16(a0, b0n, acc[2]);
    acc[0] = wmma_f16(a1, b1r, acc[0]);
    acc[1] = wmma_f16(a1, b1i, acc[1]);
    acc[2] = wmma_f16(a1, b1n, acc[2]);
  }

  int u = u16 + lane15;
  float bh_r = bhh[u];
  float bh_i = bhh[1024 + u];
  float bh_n = bhh[2048 + u];
#pragma unroll
  for (int v = 0; v < 8; ++v) {
    int b = rowM + half * 8 + v;
    const float* gs = gsum_t + (size_t)b * 3072;
    float hr = acc[0][v] + bh_r;
    float hi = acc[1][v] + bh_i;
    float hn = acc[2][v] + bh_n;
    float r  = 1.f / (1.f + __expf(-(gs[u]        + hr)));
    float ig = 1.f / (1.f + __expf(-(gs[1024 + u] + hi)));
    float n  = tanhf(gs[2048 + u] + r * hn);
    size_t hidx = (size_t)b * 1024 + u;
    float hold = h32[hidx];
    float hnew = n + ig * (hold - n);
    h32[hidx]     = hnew;
    h16_out[hidx] = (f16)hnew;
    outs_t[hidx]  = (f16)hnew;
  }
}

// ---------------------------------------------------------------------------
// Helpers
// ---------------------------------------------------------------------------
__global__ void msrnn_cvt_f32_f16(const float* __restrict__ s, f16* __restrict__ d, int n) {
  int i = blockIdx.x * blockDim.x + threadIdx.x;
  if (i < n) d[i] = (f16)s[i];
}

// moving mean over time with window `win`, transposed from (B,S,D) to (T,B,D), f16 out.
__global__ void msrnn_build_ms(const float* __restrict__ inputs, f16* __restrict__ ms,
                               int Tm, int win) {
  int i = blockIdx.x * blockDim.x + threadIdx.x;
  int total = Tm * 32 * 256;
  if (i >= total) return;
  int t = i / (32 * 256);
  int r = i % (32 * 256);
  int b = r / 256;
  int dd = r % 256;
  float s = 0.f;
  for (int j = 0; j < win; ++j)
    s += inputs[(size_t)b * 512 * 256 + (size_t)(t + j) * 256 + dd];
  ms[i] = (f16)(s / (float)win);
}

__global__ void msrnn_zero_f32(float* __restrict__ p, int n) {
  int i = blockIdx.x * blockDim.x + threadIdx.x;
  if (i < n) p[i] = 0.f;
}
__global__ void msrnn_zero_f16(f16* __restrict__ p, int n) {
  int i = blockIdx.x * blockDim.x + threadIdx.x;
  if (i < n) p[i] = (f16)0.f;
}
__global__ void msrnn_copy_f32(const float* __restrict__ s, float* __restrict__ d, int n) {
  int i = blockIdx.x * blockDim.x + threadIdx.x;
  if (i < n) d[i] = s[i];
}

// ---------------------------------------------------------------------------
// Host orchestration
// ---------------------------------------------------------------------------
static inline size_t msrnn_align256(size_t x) { return (x + 255) & ~(size_t)255; }

extern "C" void kernel_launch(void* const* d_in, const int* in_sizes, int n_in,
                              void* d_out, int out_size, void* d_ws, size_t ws_size,
                              hipStream_t stream) {
  (void)in_sizes; (void)n_in; (void)out_size; (void)ws_size;

  const float* inputs = (const float*)d_in[0];
  const float *Wxh[3], *bxh[3], *Whh[3], *bhh[3], *Wnh[3], *bnh[3];
  for (int l = 0; l < 3; ++l) {
    int base = 1 + 6 * l;
    Wxh[l] = (const float*)d_in[base + 0];
    bxh[l] = (const float*)d_in[base + 1];
    Whh[l] = (const float*)d_in[base + 2];
    bhh[l] = (const float*)d_in[base + 3];
    Wnh[l] = (const float*)d_in[base + 4];
    bnh[l] = (const float*)d_in[base + 5];
  }
  const float* Wout = (const float*)d_in[19];
  const float* bout = (const float*)d_in[20];
  float* out = (float*)d_out;

  // ---- workspace carve-out (all regions 256B aligned) ----
  char* ws = (char*)d_ws;
  size_t off = 0;
  auto alloc = [&](size_t bytes) -> char* {
    char* p = ws + off;
    off = msrnn_align256(off + bytes);
    return p;
  };
  f16 *Wxh16[3], *Whh16[3], *Wnh16[3];
  for (int l = 0; l < 3; ++l) {
    int din = (l == 0) ? 256 : 1024;
    Wxh16[l] = (f16*)alloc((size_t)3072 * din * 2);
    Whh16[l] = (f16*)alloc((size_t)3072 * 1024 * 2);
    Wnh16[l] = (f16*)alloc((size_t)3072 * 256 * 2);
  }
  f16* Wout16 = (f16*)alloc((size_t)256 * 1024 * 2);
  f16* x16    = (f16*)alloc((size_t)512 * 32 * 256 * 2);
  f16* ms16   = (f16*)alloc((size_t)512 * 32 * 256 * 2);
  float* gsum = (float*)alloc((size_t)512 * 32 * 3072 * 4);
  f16* outs16[3];
  for (int l = 0; l < 3; ++l) outs16[l] = (f16*)alloc((size_t)512 * 32 * 1024 * 2);
  float* h32  = (float*)alloc((size_t)32 * 1024 * 4);
  f16* h16buf[2];
  h16buf[0] = (f16*)alloc((size_t)32 * 1024 * 2);
  h16buf[1] = (f16*)alloc((size_t)32 * 1024 * 2);

  auto cvt = [&](const float* s, f16* d, int n) {
    msrnn_cvt_f32_f16<<<(n + 255) / 256, 256, 0, stream>>>(s, d, n);
  };

  // ---- weight conversion to f16 (once per launch) ----
  for (int l = 0; l < 3; ++l) {
    int din = (l == 0) ? 256 : 1024;
    cvt(Wxh[l], Wxh16[l], 3072 * din);
    cvt(Whh[l], Whh16[l], 3072 * 1024);
    cvt(Wnh[l], Wnh16[l], 3072 * 256);
  }
  cvt(Wout, Wout16, 256 * 1024);

  // x transposed (T,B,D) in f16 == moving mean with window 1
  {
    int tot = 512 * 32 * 256;
    msrnn_build_ms<<<(tot + 255) / 256, 256, 0, stream>>>(inputs, x16, 512, 1);
  }

  size_t yoff = 0;
  size_t hoff = (size_t)(512 + 511 + 509) * 32 * 256;

  int Tin = 512;
  int din = 256;
  const f16* layer_in = x16;

  for (int l = 0; l < 3; ++l) {
    int win = 1 << l;
    int Tm = 512 - win + 1;     // 512, 511, 509
    int skip = Tin - Tm;
    const f16* Agx = layer_in + (size_t)skip * 32 * din;

    const f16* msp = x16;
    if (l > 0) {
      int tot = Tm * 32 * 256;
      msrnn_build_ms<<<(tot + 255) / 256, 256, 0, stream>>>(inputs, ms16, Tm, win);
      msp = ms16;
    }

    // gsum = xs @ Wxh^T + (bxh + bnh)   [M = Tm*32, N = 3072, K = din]
    msrnn_gemm_f16<<<dim3(12, Tm), 256, 0, stream>>>(
        Agx, din, Wxh16[l], din, bxh[l], bnh[l], gsum, 3072, 0);
    // gsum += ms @ Wnh^T                 [K = 256]
    msrnn_gemm_f16<<<dim3(12, Tm), 256, 0, stream>>>(
        msp, 256, Wnh16[l], 256, nullptr, nullptr, gsum, 3072, 1);

    // h0 = 0
    msrnn_zero_f32<<<(32768 + 255) / 256, 256, 0, stream>>>(h32, 32768);
    msrnn_zero_f16<<<(32768 + 255) / 256, 256, 0, stream>>>(h16buf[0], 32768);

    // sequential scan (one launch per timestep; graph-captured by harness)
    for (int t = 0; t < Tm; ++t) {
      msrnn_gru_step<<<16, 256, 0, stream>>>(
          h16buf[t & 1], Whh16[l], bhh[l],
          gsum + (size_t)t * 32 * 3072,
          h32, h16buf[(t + 1) & 1],
          outs16[l] + (size_t)t * 32 * 1024);
    }

    // h_last -> d_out
    msrnn_copy_f32<<<(32768 + 255) / 256, 256, 0, stream>>>(
        h32, out + hoff + (size_t)l * 32 * 1024, 32768);

    // y = outs @ Wout^T + bout  [M = Tm*32, N = 256, K = 1024] -> d_out
    msrnn_gemm_f16<<<dim3(1, Tm), 256, 0, stream>>>(
        outs16[l], 1024, Wout16, 1024, bout, nullptr, out + yoff, 256, 0);

    yoff += (size_t)Tm * 32 * 256;
    layer_in = outs16[l];
    din = 1024;
    Tin = Tm;
  }
}
